// Mymodel_6468220748497
// MI455X (gfx1250) — matrix operand order repaired and checked by reading, not compiled, not run on hardware
//
#include <hip/hip_runtime.h>
#include <hip/hip_bf16.h>
#include <cstdint>

// ---------------------------------------------------------------------------
// Seq2seq (biLSTM encoder + Bahdanau attention decoder + vocab NLL) on gfx1250
// Core matmul primitive: v_wmma_f32_16x16x32_f16 (f16 in, f32 accumulate).
// All GEMM operands are pre-converted to f16 once; hot loops are pure
// b128 loads + WMMA. Recurrent state: cell state in VGPRs, hidden in LDS f16.
// Logits GEMM stages B tiles via global_load_async_to_lds_b128 (ASYNCcnt).
// ---------------------------------------------------------------------------

typedef __attribute__((ext_vector_type(16))) _Float16 v16h;
typedef __attribute__((ext_vector_type(8)))  _Float16 v8h;
typedef __attribute__((ext_vector_type(8)))  float    v8f;

#define S_LEN 400
#define B_N   32
#define E_N   256
#define H_N   512
#define V_N   50000
#define T_N   100
#define G4H   2048   // 4*H

__device__ __forceinline__ v8f wmma16(v16h a, v16h b, v8f c) {
  return __builtin_amdgcn_wmma_f32_16x16x32_f16(false, a, false, b, (short)0, c,
                                                false, false);
}

__device__ __forceinline__ float sigm(float x) { return 1.0f / (1.0f + __expf(-x)); }

__device__ __forceinline__ v16h mkv16(v8h lo, v8h hv) {
  return __builtin_shufflevector(lo, hv, 0, 1, 2, 3, 4, 5, 6, 7, 8, 9, 10, 11, 12, 13, 14, 15);
}

// --- A fragment (16x32 f16). ISA §7.12.2: lanes 0-15 (row M=l) hold K 0..7
// then 16..23; lanes 16-31 hold K 8..15 then 24..31. Works for LDS or global.
__device__ __forceinline__ v16h loadA_f16(const _Float16* A, int lda, int lane) {
  int l = lane & 15, hi = (lane >> 4) & 1;
  const _Float16* r = A + (size_t)l * lda;
  return mkv16(*(const v8h*)(r + hi * 8), *(const v8h*)(r + 16 + hi * 8));
}

// --- B fragment (32x16) for C = X @ W^T: B[k][n] = W[n][k]; lane layout
// N = lane&15, K = (lane>>4)*8 + j then 16 + (lane>>4)*8 + j (matches A).
__device__ __forceinline__ v16h loadB_f16(const _Float16* Wt, int ldw, int lane) {
  int l = lane & 15, hi = (lane >> 4) & 1;
  const _Float16* r = Wt + (size_t)l * ldw + hi * 8;
  return mkv16(*(const v8h*)r, *(const v8h*)(r + 16));
}

// --- gfx1250 async copy: 16B global -> LDS, tracked by ASYNCcnt.
__device__ __forceinline__ void async_b128(uint32_t lds_off, const void* g) {
  asm volatile("global_load_async_to_lds_b128 %0, %1, off"
               :: "v"(lds_off), "v"(g) : "memory");
}
__device__ __forceinline__ void wait_async0() {
  asm volatile("s_wait_asynccnt 0" ::: "memory");
}

// ===========================================================================
// Kernel 0: one-shot f32 -> f16 conversion (weights + embeddings)
// ===========================================================================
__global__ void cvt_f16(const float* __restrict__ in, _Float16* __restrict__ out, int n) {
  for (int i = blockIdx.x * blockDim.x + threadIdx.x; i < n; i += gridDim.x * blockDim.x)
    out[i] = (_Float16)in[i];
}

// ===========================================================================
// Kernel 1: encoder input projection  xW[dir] = x @ Wih[dir]^T + bias
// ===========================================================================
__global__ void enc_inproj(const _Float16* __restrict__ x16,
                           const _Float16* __restrict__ Wf16, const float* __restrict__ b_f,
                           const _Float16* __restrict__ Wb16, const float* __restrict__ b_b,
                           _Float16* __restrict__ xWf, _Float16* __restrict__ xWb) {
  int lane = threadIdx.x & 31;
  int w    = threadIdx.x >> 5;          // 0..7
  int mt   = blockIdx.x;                // 0..799
  int nt   = blockIdx.y * 8 + w;        // 0..127
  int dir  = blockIdx.z;
  const _Float16* W = dir ? Wb16 : Wf16;
  const float* bias = dir ? b_b : b_f;
  _Float16* out     = dir ? xWb : xWf;

  v8f acc = {};
#pragma unroll
  for (int k0 = 0; k0 < E_N; k0 += 32) {
    v16h a = loadA_f16(x16 + (size_t)mt * 16 * E_N + k0, E_N, lane);
    v16h b = loadB_f16(W + (size_t)nt * 16 * E_N + k0, E_N, lane);
    acc = wmma16(a, b, acc);
  }
  int l = lane & 15, hi = (lane >> 4) & 1;
  float bv = bias[nt * 16 + l];
#pragma unroll
  for (int r = 0; r < 8; ++r) {
    int m = mt * 16 + hi * 8 + r;
    out[(size_t)m * G4H + nt * 16 + l] = (_Float16)(acc[r] + bv);
  }
}

// ===========================================================================
// Kernel 2: encoder LSTM recurrence (persistent; 1 block per direction).
//   Wave w owns hidden columns [16w,16w+16) across all 4 gates; cell state in
//   registers for all 400 steps; h in LDS f16; h_s written as f16.
// ===========================================================================
__global__ void __launch_bounds__(1024) enc_rec(const _Float16* __restrict__ xWf,
                                                const _Float16* __restrict__ xWb,
                                                const _Float16* __restrict__ Whhf16,
                                                const _Float16* __restrict__ Whhb16,
                                                _Float16* __restrict__ hs16) {
  __shared__ _Float16 hF16[B_N * H_N];   // 32KB
  int lane = threadIdx.x & 31, w = threadIdx.x >> 5;
  int l = lane & 15, hi = (lane >> 4) & 1;
  int dir = blockIdx.x;
  const _Float16* xW  = dir ? xWb : xWf;
  const _Float16* Whh = dir ? Whhb16 : Whhf16;

  for (int i = threadIdx.x; i < B_N * H_N; i += 1024) hF16[i] = (_Float16)0.f;
  v8f cst[2];
#pragma unroll
  for (int mt = 0; mt < 2; ++mt)
#pragma unroll
    for (int r = 0; r < 8; ++r) cst[mt][r] = 0.f;
  __syncthreads();

  for (int step = 0; step < S_LEN; ++step) {
    int s = dir ? (S_LEN - 1 - step) : step;
    v8f acc[4][2];
#pragma unroll
    for (int g = 0; g < 4; ++g)
#pragma unroll
      for (int mt = 0; mt < 2; ++mt)
#pragma unroll
        for (int r = 0; r < 8; ++r) {
          int m = mt * 16 + hi * 8 + r;
          acc[g][mt][r] = (float)xW[((size_t)s * B_N + m) * G4H + g * H_N + w * 16 + l];
        }
    for (int kk = 0; kk < 16; ++kk) {
      int k0 = kk * 32;
      v16h a0 = loadA_f16(hF16 + k0, H_N, lane);
      v16h a1 = loadA_f16(hF16 + 16 * H_N + k0, H_N, lane);
#pragma unroll
      for (int g = 0; g < 4; ++g) {
        v16h b = loadB_f16(Whh + ((size_t)(g * H_N + w * 16)) * H_N + k0, H_N, lane);
        acc[g][0] = wmma16(a0, b, acc[g][0]);
        acc[g][1] = wmma16(a1, b, acc[g][1]);
      }
    }
    __syncthreads();  // old-h reads complete
#pragma unroll
    for (int mt = 0; mt < 2; ++mt) {
#pragma unroll
      for (int r = 0; r < 8; ++r) {
        float iv = sigm(acc[0][mt][r]);
        float fv = sigm(acc[1][mt][r]);
        float gv = tanhf(acc[2][mt][r]);
        float ov = sigm(acc[3][mt][r]);
        float c  = fv * cst[mt][r] + iv * gv;
        float h  = ov * tanhf(c);
        cst[mt][r] = c;
        int m = mt * 16 + hi * 8 + r;
        int col = w * 16 + l;
        hF16[m * H_N + col] = (_Float16)h;
        hs16[((size_t)s * B_N + m) * (2 * H_N) + dir * H_N + col] = (_Float16)h;
      }
    }
    __syncthreads();
  }
}

// ===========================================================================
// Kernel 3: wa_hs = h_s @ Wa^T   ([12800,1024] x [512,1024]^T), f32 out
// ===========================================================================
__global__ void gemm_wahs(const _Float16* __restrict__ hs16, const _Float16* __restrict__ Wa16,
                          float* __restrict__ wa_hs) {
  int lane = threadIdx.x & 31, w = threadIdx.x >> 5;
  int mt = blockIdx.x;
  int nt = blockIdx.y * 8 + w;
  v8f acc = {};
  for (int k0 = 0; k0 < 2 * H_N; k0 += 32) {
    v16h a = loadA_f16(hs16 + (size_t)mt * 16 * (2 * H_N) + k0, 2 * H_N, lane);
    v16h b = loadB_f16(Wa16 + (size_t)nt * 16 * (2 * H_N) + k0, 2 * H_N, lane);
    acc = wmma16(a, b, acc);
  }
  int l = lane & 15, hi = (lane >> 4) & 1;
#pragma unroll
  for (int r = 0; r < 8; ++r)
    wa_hs[(size_t)(mt * 16 + hi * 8 + r) * H_N + nt * 16 + l] = acc[r];
}

// ===========================================================================
// Kernel 4: attention decoder (persistent single block; 100 sequential steps)
// ===========================================================================
__global__ void __launch_bounds__(1024) decoder(
    const _Float16* __restrict__ hs16, const float* __restrict__ wa_hs,
    const float* __restrict__ s_mask, const _Float16* __restrict__ Va16,
    const float* __restrict__ Ua, const _Float16* __restrict__ dWih16,
    const _Float16* __restrict__ dWhh16, const float* __restrict__ dec_b,
    float* __restrict__ sVa, float* __restrict__ scores, float* __restrict__ alpha,
    float* __restrict__ ctx, _Float16* __restrict__ ctx16, _Float16* __restrict__ s16) {
  __shared__ _Float16 sF16[B_N * H_N];   // 32KB decoder hidden state
  int lane = threadIdx.x & 31, w = threadIdx.x >> 5;
  int l = lane & 15, hi = (lane >> 4) & 1;
  for (int i = threadIdx.x; i < B_N * H_N; i += 1024) sF16[i] = (_Float16)0.f;
  v8f yst[2];
#pragma unroll
  for (int mt = 0; mt < 2; ++mt)
#pragma unroll
    for (int r = 0; r < 8; ++r) yst[mt][r] = 0.f;
  __syncthreads();

  for (int t = 0; t < T_N; ++t) {
    // --- 1. sVa = s @ Va^T (64 tiles; 2 per wave)
#pragma unroll
    for (int q = 0; q < 2; ++q) {
      int tid = w * 2 + q;
      int mt = tid >> 5, nt = tid & 31;
      v8f acc = {};
      for (int k0 = 0; k0 < H_N; k0 += 32) {
        v16h a = loadA_f16(sF16 + mt * 16 * H_N + k0, H_N, lane);
        v16h b = loadB_f16(Va16 + (size_t)nt * 16 * H_N + k0, H_N, lane);
        acc = wmma16(a, b, acc);
      }
#pragma unroll
      for (int r = 0; r < 8; ++r)
        sVa[(mt * 16 + hi * 8 + r) * H_N + nt * 16 + l] = acc[r];
    }
    __syncthreads();
    // --- 2. scores[s,b] = Ua . tanh(wa_hs[s,b,:] + sVa[b,:])  (then *mask)
    for (int p = threadIdx.x; p < S_LEN * B_N; p += 1024) {
      int b = p & 31;
      const float* wrow = wa_hs + (size_t)p * H_N;
      const float* srow = sVa + b * H_N;
      float acc = 0.f;
      for (int h = 0; h < H_N; ++h) acc += Ua[h] * tanhf(wrow[h] + srow[h]);
      scores[p] = acc * s_mask[p];
    }
    __syncthreads();
    // --- 3. softmax over source positions (wave w -> batch column w)
    {
      int b = w;
      float mx = -1e30f;
      for (int sp = lane; sp < S_LEN; sp += 32) mx = fmaxf(mx, scores[sp * B_N + b]);
      for (int off = 16; off; off >>= 1) mx = fmaxf(mx, __shfl_xor(mx, off, 32));
      float sum = 0.f;
      for (int sp = lane; sp < S_LEN; sp += 32) sum += __expf(scores[sp * B_N + b] - mx);
      for (int off = 16; off; off >>= 1) sum += __shfl_xor(sum, off, 32);
      float inv = 1.f / sum;
      for (int sp = lane; sp < S_LEN; sp += 32)
        alpha[sp * B_N + b] = __expf(scores[sp * B_N + b] - mx) * inv;
    }
    __syncthreads();
    // --- 4. ctx[b,d] = sum_s alpha[s,b] * h_s[s,b,d]
    for (int p = threadIdx.x; p < B_N * 2 * H_N; p += 1024) {
      int b = p >> 10, d = p & 1023;
      float acc = 0.f;
      for (int sp = 0; sp < S_LEN; ++sp)
        acc += alpha[sp * B_N + b] * (float)hs16[((size_t)sp * B_N + b) * (2 * H_N) + d];
      ctx[p] = acc;
      ctx16[p] = (_Float16)acc;
    }
    __syncthreads();
    // --- 5. LSTMCell gates = ctx@Wih^T + s@Whh^T + b
    v8f acc[4][2];
#pragma unroll
    for (int g = 0; g < 4; ++g) {
      float bv = dec_b[g * H_N + w * 16 + l];
#pragma unroll
      for (int mt = 0; mt < 2; ++mt)
#pragma unroll
        for (int r = 0; r < 8; ++r) acc[g][mt][r] = bv;
    }
    for (int kk = 0; kk < 32; ++kk) {  // ctx, K = 1024
      int k0 = kk * 32;
      v16h a0 = loadA_f16(ctx16 + k0, 2 * H_N, lane);
      v16h a1 = loadA_f16(ctx16 + 16 * (2 * H_N) + k0, 2 * H_N, lane);
#pragma unroll
      for (int g = 0; g < 4; ++g) {
        v16h b = loadB_f16(dWih16 + ((size_t)(g * H_N + w * 16)) * (2 * H_N) + k0,
                           2 * H_N, lane);
        acc[g][0] = wmma16(a0, b, acc[g][0]);
        acc[g][1] = wmma16(a1, b, acc[g][1]);
      }
    }
    for (int kk = 0; kk < 16; ++kk) {  // s, K = 512
      int k0 = kk * 32;
      v16h a0 = loadA_f16(sF16 + k0, H_N, lane);
      v16h a1 = loadA_f16(sF16 + 16 * H_N + k0, H_N, lane);
#pragma unroll
      for (int g = 0; g < 4; ++g) {
        v16h b = loadB_f16(dWhh16 + ((size_t)(g * H_N + w * 16)) * H_N + k0, H_N, lane);
        acc[g][0] = wmma16(a0, b, acc[g][0]);
        acc[g][1] = wmma16(a1, b, acc[g][1]);
      }
    }
    __syncthreads();
#pragma unroll
    for (int mt = 0; mt < 2; ++mt) {
#pragma unroll
      for (int r = 0; r < 8; ++r) {
        float iv = sigm(acc[0][mt][r]);
        float fv = sigm(acc[1][mt][r]);
        float gv = tanhf(acc[2][mt][r]);
        float ov = sigm(acc[3][mt][r]);
        float y  = fv * yst[mt][r] + iv * gv;
        float s2 = ov * tanhf(y);
        yst[mt][r] = y;
        int m = mt * 16 + hi * 8 + r;
        int col = w * 16 + l;
        sF16[m * H_N + col] = (_Float16)s2;
        s16[m * H_N + col] = (_Float16)s2;
      }
    }
    __syncthreads();
  }
}

// ===========================================================================
// Kernel 5: logits = s_fin @ Wb^T + ctx_fin @ Vb^T   ([32,50000])
//   B tiles staged into LDS via global_load_async_to_lds_b128 (ASYNCcnt),
//   then consumed twice (both M-tiles) from LDS.
// ===========================================================================
__global__ void __launch_bounds__(256) gemm_logits(
    const _Float16* __restrict__ s16, const _Float16* __restrict__ ctx16,
    const _Float16* __restrict__ Wb16, const _Float16* __restrict__ Vb16,
    float* __restrict__ logits) {
  __shared__ _Float16 Bt[4 * 16 * 32];   // 4KB staging: 4 n-tiles x 16 rows x 32 halves
  const int NT = V_N / 16;               // 3125
  int tid = threadIdx.x;
  int lane = tid & 31, w = tid >> 5;
  int mt = w & 1;
  int qt = w >> 1;                       // consuming tile within block (wave-uniform)
  int nt = blockIdx.x * 4 + qt;
  bool valid = nt < NT;
  // staging role: thread stages one 16B chunk per k-chunk (wave-uniform tile)
  int sq = tid >> 6, srow = (tid >> 2) & 15, sseg = tid & 3;
  int gn = blockIdx.x * 4 + sq;
  bool sval = gn < NT;
  uint32_t ldst = (uint32_t)(uintptr_t)&Bt[sq * 512 + srow * 32 + sseg * 8];

  v8f acc = {};
  for (int k0 = 0; k0 < H_N; k0 += 32) {
    if (sval) async_b128(ldst, Wb16 + ((size_t)gn * 16 + srow) * H_N + k0 + sseg * 8);
    wait_async0();
    __syncthreads();
    if (valid) {
      v16h a = loadA_f16(s16 + mt * 16 * H_N + k0, H_N, lane);
      v16h b = loadB_f16(&Bt[qt * 512], 32, lane);
      acc = wmma16(a, b, acc);
    }
    __syncthreads();
  }
  for (int k0 = 0; k0 < 2 * H_N; k0 += 32) {
    if (sval) async_b128(ldst, Vb16 + ((size_t)gn * 16 + srow) * (2 * H_N) + k0 + sseg * 8);
    wait_async0();
    __syncthreads();
    if (valid) {
      v16h a = loadA_f16(ctx16 + mt * 16 * (2 * H_N) + k0, 2 * H_N, lane);
      v16h b = loadB_f16(&Bt[qt * 512], 32, lane);
      acc = wmma16(a, b, acc);
    }
    __syncthreads();
  }
  if (valid) {
    int l = lane & 15, hi = (lane >> 4) & 1;
#pragma unroll
    for (int r = 0; r < 8; ++r)
      logits[(size_t)(mt * 16 + hi * 8 + r) * V_N + nt * 16 + l] = acc[r];
  }
}

// ===========================================================================
// Kernel 6: per-batch log-softmax + masked NLL mean  -> loss[32]
// ===========================================================================
__global__ void loss_kernel(const float* __restrict__ logits, const int* __restrict__ t_idx,
                            const float* __restrict__ t_mask, float* __restrict__ out) {
  int b = blockIdx.x;
  const float* row = logits + (size_t)b * V_N;
  __shared__ float red[256];
  float mx = -1e30f;
  for (int v = threadIdx.x; v < V_N; v += 256) mx = fmaxf(mx, row[v]);
  red[threadIdx.x] = mx; __syncthreads();
  for (int o = 128; o; o >>= 1) { if (threadIdx.x < o) red[threadIdx.x] = fmaxf(red[threadIdx.x], red[threadIdx.x + o]); __syncthreads(); }
  mx = red[0]; __syncthreads();
  float sum = 0.f;
  for (int v = threadIdx.x; v < V_N; v += 256) sum += __expf(row[v] - mx);
  red[threadIdx.x] = sum; __syncthreads();
  for (int o = 128; o; o >>= 1) { if (threadIdx.x < o) red[threadIdx.x] += red[threadIdx.x + o]; __syncthreads(); }
  float lse = mx + logf(red[0]); __syncthreads();
  float nll = 0.f, msum = 0.f;
  for (int t = threadIdx.x; t < T_N; t += 256) {
    int idx = t_idx[t * B_N + b];
    float m = t_mask[t * B_N + b];
    nll += (lse - row[idx]) * m;
    msum += m;
  }
  red[threadIdx.x] = nll; __syncthreads();
  for (int o = 128; o; o >>= 1) { if (threadIdx.x < o) red[threadIdx.x] += red[threadIdx.x + o]; __syncthreads(); }
  float tn = red[0]; __syncthreads();
  red[threadIdx.x] = msum; __syncthreads();
  for (int o = 128; o; o >>= 1) { if (threadIdx.x < o) red[threadIdx.x] += red[threadIdx.x + o]; __syncthreads(); }
  if (threadIdx.x == 0) out[b] = tn / red[0];
}

// ===========================================================================
extern "C" void kernel_launch(void* const* d_in, const int* in_sizes, int n_in,
                              void* d_out, int out_size, void* d_ws, size_t ws_size,
                              hipStream_t stream) {
  const float* x        = (const float*)d_in[0];
  const float* s_mask   = (const float*)d_in[1];
  const int*   t_idx    = (const int*)  d_in[2];
  const float* t_mask   = (const float*)d_in[3];
  const float* Wih_f    = (const float*)d_in[4];
  const float* Whh_f    = (const float*)d_in[5];
  const float* b_f      = (const float*)d_in[6];
  const float* Wih_b    = (const float*)d_in[7];
  const float* Whh_b    = (const float*)d_in[8];
  const float* b_b      = (const float*)d_in[9];
  const float* dec_Wih  = (const float*)d_in[10];
  const float* dec_Whh  = (const float*)d_in[11];
  const float* dec_b    = (const float*)d_in[12];
  const float* Wa       = (const float*)d_in[13];
  const float* Va       = (const float*)d_in[14];
  const float* Ua       = (const float*)d_in[15];
  const float* Wb       = (const float*)d_in[16];
  const float* Vb       = (const float*)d_in[17];
  float* out = (float*)d_out;

  char* ws = (char*)d_ws;
  size_t off = 0;
  auto take = [&](size_t bytes) { char* p = ws + off; off += (bytes + 255) & ~(size_t)255; return p; };
  // f16 shadows of embeddings + all GEMM weights
  _Float16* x16    = (_Float16*)take((size_t)S_LEN * B_N * E_N * 2);
  _Float16* Wihf16 = (_Float16*)take((size_t)G4H * E_N * 2);
  _Float16* Wihb16 = (_Float16*)take((size_t)G4H * E_N * 2);
  _Float16* Whhf16 = (_Float16*)take((size_t)G4H * H_N * 2);
  _Float16* Whhb16 = (_Float16*)take((size_t)G4H * H_N * 2);
  _Float16* dWih16 = (_Float16*)take((size_t)G4H * 2 * H_N * 2);
  _Float16* dWhh16 = (_Float16*)take((size_t)G4H * H_N * 2);
  _Float16* Wa16   = (_Float16*)take((size_t)H_N * 2 * H_N * 2);
  _Float16* Va16   = (_Float16*)take((size_t)H_N * H_N * 2);
  _Float16* Wb16   = (_Float16*)take((size_t)V_N * H_N * 2);
  _Float16* Vb16   = (_Float16*)take((size_t)V_N * 2 * H_N * 2);
  // activations
  _Float16* xWf    = (_Float16*)take((size_t)S_LEN * B_N * G4H * 2);
  _Float16* xWb    = (_Float16*)take((size_t)S_LEN * B_N * G4H * 2);
  _Float16* hs16   = (_Float16*)take((size_t)S_LEN * B_N * 2 * H_N * 2);
  float*    wa_hs  = (float*)   take((size_t)S_LEN * B_N * H_N * 4);
  float*    sVa    = (float*)   take((size_t)B_N * H_N * 4);
  float*    sc     = (float*)   take((size_t)S_LEN * B_N * 4);
  float*    al     = (float*)   take((size_t)S_LEN * B_N * 4);
  float*    ctx    = (float*)   take((size_t)B_N * 2 * H_N * 4);
  _Float16* ctx16  = (_Float16*)take((size_t)B_N * 2 * H_N * 2);
  _Float16* s16    = (_Float16*)take((size_t)B_N * H_N * 2);
  float*    lg     = (float*)   take((size_t)B_N * V_N * 4);
  (void)ws_size; (void)in_sizes; (void)n_in; (void)out_size;

  // 0. one-shot f32 -> f16 conversions
  cvt_f16<<<1024, 256, 0, stream>>>(x,       x16,    S_LEN * B_N * E_N);
  cvt_f16<<<1024, 256, 0, stream>>>(Wih_f,   Wihf16, G4H * E_N);
  cvt_f16<<<1024, 256, 0, stream>>>(Wih_b,   Wihb16, G4H * E_N);
  cvt_f16<<<1024, 256, 0, stream>>>(Whh_f,   Whhf16, G4H * H_N);
  cvt_f16<<<1024, 256, 0, stream>>>(Whh_b,   Whhb16, G4H * H_N);
  cvt_f16<<<1024, 256, 0, stream>>>(dec_Wih, dWih16, G4H * 2 * H_N);
  cvt_f16<<<1024, 256, 0, stream>>>(dec_Whh, dWhh16, G4H * H_N);
  cvt_f16<<<1024, 256, 0, stream>>>(Wa,      Wa16,   H_N * 2 * H_N);
  cvt_f16<<<1024, 256, 0, stream>>>(Va,      Va16,   H_N * H_N);
  cvt_f16<<<4096, 256, 0, stream>>>(Wb,      Wb16,   V_N * H_N);
  cvt_f16<<<4096, 256, 0, stream>>>(Vb,      Vb16,   V_N * 2 * H_N);

  // 1. encoder input projections (both directions)
  enc_inproj<<<dim3(S_LEN * B_N / 16, G4H / (16 * 8), 2), 256, 0, stream>>>(
      x16, Wihf16, b_f, Wihb16, b_b, xWf, xWb);
  // 2. persistent biLSTM recurrence (1 block/direction)
  enc_rec<<<2, 1024, 0, stream>>>(xWf, xWb, Whhf16, Whhb16, hs16);
  // 3. hoisted attention projection
  gemm_wahs<<<dim3(S_LEN * B_N / 16, 4), 256, 0, stream>>>(hs16, Wa16, wa_hs);
  // 4. persistent attention decoder
  decoder<<<1, 1024, 0, stream>>>(hs16, wa_hs, s_mask, Va16, Ua, dWih16, dWhh16,
                                  dec_b, sVa, sc, al, ctx, ctx16, s16);
  // 5. vocab logits (async-LDS staged B tiles)
  gemm_logits<<<(V_N / 16 + 3) / 4, 256, 0, stream>>>(s16, ctx16, Wb16, Vb16, lg);
  // 6. log-softmax + masked NLL
  loss_kernel<<<B_N, 256, 0, stream>>>(lg, t_idx, t_mask, out);
}
